// T5_71665824301626
// MI455X (gfx1250) — matrix operand order repaired and checked by reading, not compiled, not run on hardware
//
#include <hip/hip_runtime.h>

// ---------------------------------------------------------------------------
// DeBERTa disentangled-attention block for MI455X (gfx1250).
// All matmuls on v_wmma_f32_16x16x32_bf16 (fp32 accumulate), big GEMMs use the
// Tensor Data Mover (tensor_load_to_lds) with LDS double buffering. wave32.
// ---------------------------------------------------------------------------

typedef __bf16 bf16;
typedef __attribute__((ext_vector_type(16))) __bf16 bf16x16;
typedef __attribute__((ext_vector_type(8)))  float  f32x8;
typedef __attribute__((ext_vector_type(4)))  unsigned int uint32x4;
typedef __attribute__((ext_vector_type(8)))  int int32x8;
typedef __attribute__((ext_vector_type(4)))  int int32x4;

#define SEQ 512
#define BB  16
#define HH  1024
#define NHH 16
#define HDD 64
#define LN_EPS 1e-7f
#define ATT_SCALE 0.07216878364870322f   // 1/sqrt(3*64)

// ---- workspace layout (bytes) ---------------------------------------------
#define SZ_ACT   ((size_t)SEQ * BB * HH * 2)   // 16 MiB  (8192 x 1024 bf16)
#define SZ_W     ((size_t)HH * HH * 2)         // 2 MiB
#define SZ_REL   ((size_t)64 * HH * 2)         // 128 KiB (padded to 64 rows)

#define OFF_XNB  ((size_t)0)                   // reused for proj-out later
#define OFF_WQB  (OFF_XNB + SZ_ACT)
#define OFF_WKB  (OFF_WQB + SZ_W)
#define OFF_WVB  (OFF_WKB + SZ_W)
#define OFF_WOB  (OFF_WVB + SZ_W)
#define OFF_RELB (OFF_WOB + SZ_W)
#define OFF_QS   (OFF_RELB + SZ_REL)
#define OFF_KB   (OFF_QS + SZ_ACT)
#define OFF_VB   (OFF_KB + SZ_ACT)
#define OFF_QPS  (OFF_VB + SZ_ACT)
#define OFF_KPB  (OFF_QPS + SZ_REL)
#define OFF_CPB  (OFF_KPB + SZ_REL)
#define OFF_PCB  (OFF_CPB + SZ_ACT)
#define OFF_CTX  (OFF_PCB + SZ_ACT)

// ---- WMMA helpers ----------------------------------------------------------
__device__ __forceinline__ f32x8 wmma_bf16(bf16x16 a, bf16x16 b, f32x8 c) {
  return __builtin_amdgcn_wmma_f32_16x16x32_bf16(
      false, a, false, b, (short)0, c, false, false);
}

// A-matrix 16x32 bf16 lane->element map (ISA 7.12.2):
// lane L holds row M=L&15; K set = {b..b+7} U {b+16..b+23}, b = (L>>4)*8.
__device__ __forceinline__ int kmapA(int lane, int e) {
  return ((lane >> 4) << 3) + e + ((e >= 8) ? 8 : 0);
}

__device__ __forceinline__ bf16x16 load_A_frag(const bf16* base, int ld, int lane) {
  const bf16* p = base + (size_t)(lane & 15) * (size_t)ld;
  bf16x16 a;
#pragma unroll
  for (int e = 0; e < 16; ++e) a[e] = p[kmapA(lane, e)];
  return a;
}

// ---- Tensor Data Mover: 2D tile global->LDS (D# per CDNA5 ISA ch.8) --------
__device__ __forceinline__ unsigned lds_addr_of(const void* p) {
  return (unsigned)(unsigned long long)(size_t)p;   // LDS aperture low 32 bits
}

// Loads a [tile_d1 rows x tile_d0 elems] bf16 tile (row stride = stride elems)
// from gaddr into contiguous LDS at lds_off. 2D: D# groups 2/3 zero.
// NOTE: this toolchain's builtin is the 6-arg (clang-23 / therock) form:
//   (v4u g0, v8i g1, v4i g2, v4i g3, v8i extra, i32 cpol)
__device__ __forceinline__ void tdm_load_2d(const void* gaddr, unsigned lds_off,
                                            unsigned tile_d0, unsigned tile_d1,
                                            unsigned long long stride) {
  unsigned long long ga = (unsigned long long)gaddr;
  uint32x4 g0;
  g0[0] = 1u;                                        // count=1, no gather
  g0[1] = lds_off;                                   // lds_addr
  g0[2] = (unsigned)(ga & 0xffffffffu);              // global_addr lo
  g0[3] = (unsigned)((ga >> 32) & 0x01ffffffu) | (2u << 30);  // addr hi | type=2
  int32x8 g1;
  g1[0] = (int)(1u << 16);                           // data_size=1 -> 2 bytes
  g1[1] = (int)((tile_d0 & 0xffffu) << 16);          // tensor_dim0[15:0]
  g1[2] = (int)(((tile_d0 >> 16) & 0xffffu) | ((tile_d1 & 0xffffu) << 16));
  g1[3] = (int)(((tile_d1 >> 16) & 0xffffu) | (tile_d0 << 16));   // tile_dim0
  g1[4] = (int)(tile_d1 & 0xffffu);                  // tile_dim1 (tile_dim2=0)
  g1[5] = (int)(unsigned)(stride & 0xffffffffu);     // tensor_dim0_stride lo
  g1[6] = (int)(unsigned)((stride >> 32) & 0xffffu); // stride hi (dim1_stride=0)
  g1[7] = 0;
  int32x4 z4 = {0, 0, 0, 0};
  int32x8 z8 = {0, 0, 0, 0, 0, 0, 0, 0};
  __builtin_amdgcn_tensor_load_to_lds(g0, g1, z4, z4, z8, 0);
}

// ---- kernel 0a: LayerNorm(q) -> bf16 ---------------------------------------
__global__ __launch_bounds__(256) void ln_rows_kernel(const float* __restrict__ x,
                                                      bf16* __restrict__ y) {
  const int t = blockIdx.x;                 // token = s*B + b
  const float* xr = x + (size_t)t * HH;
  const int tid = threadIdx.x;
  float v[4], s = 0.f, s2 = 0.f;
#pragma unroll
  for (int i = 0; i < 4; ++i) {
    v[i] = xr[tid + i * 256];
    s += v[i]; s2 += v[i] * v[i];
  }
#pragma unroll
  for (int mk = 1; mk < 32; mk <<= 1) {
    s  += __shfl_xor(s, mk, 32);
    s2 += __shfl_xor(s2, mk, 32);
  }
  __shared__ float a1[8], a2[8];
  const int wid = tid >> 5, lane = tid & 31;
  if (lane == 0) { a1[wid] = s; a2[wid] = s2; }
  __syncthreads();
  float S1 = 0.f, S2 = 0.f;
#pragma unroll
  for (int w = 0; w < 8; ++w) { S1 += a1[w]; S2 += a2[w]; }
  const float mu = S1 * (1.f / HH);
  const float var = S2 * (1.f / HH) - mu * mu;
  const float rs = rsqrtf(var + LN_EPS);
#pragma unroll
  for (int i = 0; i < 4; ++i)
    y[(size_t)t * HH + tid + i * 256] = (bf16)((v[i] - mu) * rs);
}

// ---- kernel 0b: f32 -> bf16 convert ----------------------------------------
__global__ __launch_bounds__(256) void cvt_bf16_kernel(const float* __restrict__ x,
                                                       bf16* __restrict__ y, int n) {
  const int i = blockIdx.x * 256 + threadIdx.x;
  if (i < n) y[i] = (bf16)x[i];
}

// ---- kernel 0c: relative_embedding (63x1024) -> bf16 padded to 64 rows -----
__global__ __launch_bounds__(256) void relpad_kernel(const float* __restrict__ x,
                                                     bf16* __restrict__ y) {
  const int i = blockIdx.x * 256 + threadIdx.x;   // 64*1024 total
  const int r = i >> 10;
  y[i] = (r < 63) ? (bf16)x[i] : (bf16)0.0f;
}

// ---- kernel 1: TDM-tiled GEMM: C[8192x1024] = A @ W + bias, *scale -> bf16
// grid = (64, 8): 128x128 output tile per workgroup; 8 waves as 4x2, each wave
// 32 rows x 64 cols. A/W k-tiles staged in LDS by the Tensor Data Mover with
// double buffering (wave 0 drives the DMA, s_wait_tensorcnt + barrier fences).
__global__ __launch_bounds__(256) void gemm_tdm_kernel(const bf16* __restrict__ A,
                                                       const bf16* __restrict__ W,
                                                       const float* __restrict__ bias,
                                                       bf16* __restrict__ out,
                                                       float scale) {
  __shared__ bf16 Asm[2][128][32];   // [buf][row][k]   8 KiB each
  __shared__ bf16 Wsm[2][32][128];   // [buf][k][col]   8 KiB each
  const int tid = threadIdx.x, wid = tid >> 5, lane = tid & 31;
  const int wr = wid >> 1, wc = wid & 1;
  const int row0 = blockIdx.x * 128, n0 = blockIdx.y * 128;
  const int hi = lane >> 4, cl = lane & 15;
  f32x8 acc[2][4] = {};

  if (wid == 0) {
    tdm_load_2d(A + (size_t)row0 * HH, lds_addr_of(&Asm[0][0][0]), 32, 128, HH);
    tdm_load_2d(W + n0,                lds_addr_of(&Wsm[0][0][0]), 128, 32, HH);
  }
  for (int ks = 0; ks < 32; ++ks) {
    const int cur = ks & 1, nxt = cur ^ 1;
    if (wid == 0) __builtin_amdgcn_s_wait_tensorcnt(0);
    __syncthreads();                       // cur tile visible; prev reads done
    if (wid == 0 && ks + 1 < 32) {
      const int k0n = (ks + 1) * 32;
      tdm_load_2d(A + (size_t)row0 * HH + k0n, lds_addr_of(&Asm[nxt][0][0]), 32, 128, HH);
      tdm_load_2d(W + (size_t)k0n * HH + n0,   lds_addr_of(&Wsm[nxt][0][0]), 128, 32, HH);
    }
    bf16x16 af[2];
#pragma unroll
    for (int rt = 0; rt < 2; ++rt) {
      const bf16* p = &Asm[cur][wr * 32 + rt * 16 + cl][0];
      bf16x16 a;
#pragma unroll
      for (int e = 0; e < 16; ++e) a[e] = p[kmapA(lane, e)];
      af[rt] = a;
    }
#pragma unroll
    for (int ct = 0; ct < 4; ++ct) {
      bf16x16 bm;
#pragma unroll
      for (int e = 0; e < 16; ++e) bm[e] = Wsm[cur][kmapA(lane, e)][wc * 64 + ct * 16 + cl];
#pragma unroll
      for (int rt = 0; rt < 2; ++rt) acc[rt][ct] = wmma_bf16(af[rt], bm, acc[rt][ct]);
    }
  }
#pragma unroll
  for (int rt = 0; rt < 2; ++rt)
#pragma unroll
    for (int ct = 0; ct < 4; ++ct) {
      const int col = n0 + wc * 64 + ct * 16 + cl;
      const float bv = bias[col];
#pragma unroll
      for (int r = 0; r < 8; ++r) {
        const int row = row0 + wr * 32 + rt * 16 + 8 * hi + r;
        out[(size_t)row * HH + col] = (bf16)((acc[rt][ct][r] + bv) * scale);
      }
    }
}

// ---- kernel 1b: small GEMM (M=64 rel projections), direct-from-global ------
__global__ __launch_bounds__(256) void gemm_bias_kernel(const bf16* __restrict__ A,
                                                        const bf16* __restrict__ W,
                                                        const float* __restrict__ bias,
                                                        bf16* __restrict__ out,
                                                        float scale) {
  const int tid = threadIdx.x, wid = tid >> 5, lane = tid & 31;
  const int row0 = blockIdx.x * 16, n0 = wid * 128;
  const int hi = lane >> 4, cl = lane & 15;
  f32x8 acc[8] = {};
  for (int k0 = 0; k0 < HH; k0 += 32) {
    bf16x16 a = load_A_frag(A + (size_t)row0 * HH + k0, HH, lane);
    __builtin_prefetch(W + (size_t)(k0 + 32 + lane) * HH + n0, 0, 1);
#pragma unroll
    for (int nt = 0; nt < 8; ++nt) {
      bf16x16 bm;
#pragma unroll
      for (int e = 0; e < 16; ++e) bm[e] = W[(size_t)(k0 + kmapA(lane, e)) * HH + n0 + nt * 16 + cl];
      acc[nt] = wmma_bf16(a, bm, acc[nt]);
    }
  }
#pragma unroll
  for (int nt = 0; nt < 8; ++nt) {
    const int col = n0 + nt * 16 + cl;
    const float bv = bias[col];
#pragma unroll
    for (int r = 0; r < 8; ++r)
      out[(size_t)(row0 + r + 8 * hi) * HH + col] = (bf16)((acc[nt][r] + bv) * scale);
  }
}

// ---- kernel 2: per-head  out[t][h*64+n] = sum_d A[t][h*64+d] * P[n][h*64+d]
__global__ __launch_bounds__(256) void cppc_gemm_kernel(const bf16* __restrict__ A,
                                                        const bf16* __restrict__ P,
                                                        bf16* __restrict__ out) {
  const int tid = threadIdx.x, wid = tid >> 5, lane = tid & 31;
  const int row0 = blockIdx.x * 16;
  const int h = blockIdx.y * 8 + wid;
  const int hi = lane >> 4, cl = lane & 15;
  const bf16* Ab = A + (size_t)row0 * HH + h * HDD;
  bf16x16 af[2];
#pragma unroll
  for (int kd = 0; kd < 2; ++kd) af[kd] = load_A_frag(Ab + kd * 32, HH, lane);
#pragma unroll
  for (int nt = 0; nt < 4; ++nt) {
    f32x8 c = {};
#pragma unroll
    for (int kd = 0; kd < 2; ++kd) {
      bf16x16 bm;
      const bf16* pb = P + (size_t)(nt * 16 + cl) * HH + h * HDD + kd * 32;
#pragma unroll
      for (int e = 0; e < 16; ++e) bm[e] = pb[kmapA(lane, e)];
      c = wmma_bf16(af[kd], bm, c);
    }
#pragma unroll
    for (int r = 0; r < 8; ++r)
      out[(size_t)(row0 + r + 8 * hi) * HH + h * HDD + nt * 16 + cl] = (bf16)c[r];
  }
}

// ---- kernel 3: flash-style attention with cp/pc gathers --------------------
__global__ __launch_bounds__(128) void attn_kernel(const bf16* __restrict__ Q,
                                                   const bf16* __restrict__ Kp,
                                                   const bf16* __restrict__ Vp,
                                                   const bf16* __restrict__ cp,
                                                   const bf16* __restrict__ pc,
                                                   const int* __restrict__ pidx,
                                                   bf16* __restrict__ ctx) {
  const int tid = threadIdx.x, wid = tid >> 5, lane = tid & 31;
  const int hi = lane >> 4, cl = lane & 15;
  const int q0 = blockIdx.x * 64;
  const int h = blockIdx.y, b = blockIdx.z;
  const int qrb = q0 + wid * 16;

  __shared__ bf16 KT[64][32];       // K tile transposed: [d][k]
  __shared__ bf16 VT[32][64];       // V tile: [k][d]
  __shared__ bf16 PL[4][16][32];    // per-wave P relayout buffer

  const bf16* Qbase = Q + ((size_t)qrb * BB + b) * HH + h * HDD;
  bf16x16 aq[2];
#pragma unroll
  for (int kd = 0; kd < 2; ++kd) aq[kd] = load_A_frag(Qbase + kd * 32, BB * HH, lane);

  f32x8 O[4] = {};
  float m[8], l[8];
#pragma unroll
  for (int r = 0; r < 8; ++r) { m[r] = -INFINITY; l[r] = 0.f; }

  const int kend = q0 + 64;          // causal: skip fully-masked tiles
  for (int k0 = 0; k0 < kend; k0 += 32) {
    __syncthreads();
    {
      const int kr = tid >> 2;
      const int sg = (tid & 3) * 16;
      const bf16* kgp = Kp + ((size_t)(k0 + kr) * BB + b) * HH + h * HDD + sg;
      const bf16* vgp = Vp + ((size_t)(k0 + kr) * BB + b) * HH + h * HDD + sg;
      uint4 t0 = *(const uint4*)kgp;
      uint4 t1 = *(const uint4*)(kgp + 8);
      bf16 tmp[16];
      *(uint4*)&tmp[0] = t0;
      *(uint4*)&tmp[8] = t1;
#pragma unroll
      for (int j = 0; j < 16; ++j) KT[sg + j][kr] = tmp[j];
      *(uint4*)&VT[kr][sg]     = *(const uint4*)vgp;
      *(uint4*)&VT[kr][sg + 8] = *(const uint4*)(vgp + 8);
    }
    __syncthreads();

    f32x8 sc[2];
#pragma unroll
    for (int nt = 0; nt < 2; ++nt) {
      f32x8 c = {};
#pragma unroll
      for (int kd = 0; kd < 2; ++kd) {
        bf16x16 bm;
#pragma unroll
        for (int e = 0; e < 16; ++e) bm[e] = KT[kd * 32 + kmapA(lane, e)][nt * 16 + cl];
        c = wmma_bf16(aq[kd], bm, c);
      }
      sc[nt] = c;
    }

#pragma unroll
    for (int nt = 0; nt < 2; ++nt) {
      const int kc = k0 + nt * 16 + cl;
#pragma unroll
      for (int r = 0; r < 8; ++r) {
        const int qr = qrb + 8 * hi + r;
        float v = sc[nt][r];
        const int id = pidx[qr * SEQ + kc];
        v += (float)cp[((size_t)qr * BB + b) * HH + h * HDD + id];
        v += (float)pc[((size_t)kc * BB + b) * HH + h * HDD + id];
        if (kc > qr) v = -INFINITY;
        sc[nt][r] = v;
      }
    }

#pragma unroll
    for (int r = 0; r < 8; ++r) {
      float mt = fmaxf(sc[0][r], sc[1][r]);
#pragma unroll
      for (int mk = 1; mk < 16; mk <<= 1) mt = fmaxf(mt, __shfl_xor(mt, mk, 32));
      const float mn = fmaxf(m[r], mt);
      const float al = __expf(m[r] - mn);
      m[r] = mn;
      const float p0 = __expf(sc[0][r] - mn);
      const float p1 = __expf(sc[1][r] - mn);
      float rsum = p0 + p1;
#pragma unroll
      for (int mk = 1; mk < 16; mk <<= 1) rsum += __shfl_xor(rsum, mk, 32);
      l[r] = l[r] * al + rsum;
      PL[wid][r + 8 * hi][cl]      = (bf16)p0;
      PL[wid][r + 8 * hi][16 + cl] = (bf16)p1;
#pragma unroll
      for (int t = 0; t < 4; ++t) O[t][r] *= al;
    }

    asm volatile("s_wait_dscnt 0x0" ::: "memory");

    bf16x16 ap;
    {
      const bf16* pp = &PL[wid][cl][0];
#pragma unroll
      for (int e = 0; e < 16; ++e) ap[e] = pp[kmapA(lane, e)];
    }
#pragma unroll
    for (int t = 0; t < 4; ++t) {
      bf16x16 bv;
#pragma unroll
      for (int e = 0; e < 16; ++e) bv[e] = VT[kmapA(lane, e)][t * 16 + cl];
      O[t] = wmma_bf16(ap, bv, O[t]);
    }
  }

#pragma unroll
  for (int t = 0; t < 4; ++t) {
#pragma unroll
    for (int r = 0; r < 8; ++r) {
      const int qr = qrb + 8 * hi + r;
      const int d = t * 16 + cl;
      ctx[((size_t)qr * BB + b) * HH + h * HDD + d] = (bf16)(O[t][r] / l[r]);
    }
  }
}

// ---- kernel 4: row LN with affine: out = LN(x) * g + b  (bf16 in, f32 out) -
__global__ __launch_bounds__(256) void ln_affine_kernel(const bf16* __restrict__ x,
                                                        const float* __restrict__ g,
                                                        const float* __restrict__ bb,
                                                        float* __restrict__ y) {
  const int t = blockIdx.x;
  const bf16* xr = x + (size_t)t * HH;
  const int tid = threadIdx.x;
  float v[4], s = 0.f, s2 = 0.f;
#pragma unroll
  for (int i = 0; i < 4; ++i) {
    v[i] = (float)xr[tid + i * 256];
    s += v[i]; s2 += v[i] * v[i];
  }
#pragma unroll
  for (int mk = 1; mk < 32; mk <<= 1) {
    s  += __shfl_xor(s, mk, 32);
    s2 += __shfl_xor(s2, mk, 32);
  }
  __shared__ float a1[8], a2[8];
  const int wid = tid >> 5, lane = tid & 31;
  if (lane == 0) { a1[wid] = s; a2[wid] = s2; }
  __syncthreads();
  float S1 = 0.f, S2 = 0.f;
#pragma unroll
  for (int w = 0; w < 8; ++w) { S1 += a1[w]; S2 += a2[w]; }
  const float mu = S1 * (1.f / HH);
  const float var = S2 * (1.f / HH) - mu * mu;
  const float rs = rsqrtf(var + LN_EPS);
#pragma unroll
  for (int i = 0; i < 4; ++i) {
    const int col = tid + i * 256;
    y[(size_t)t * HH + col] = (v[i] - mu) * rs * g[col] + bb[col];
  }
}

// ---------------------------------------------------------------------------
extern "C" void kernel_launch(void* const* d_in, const int* in_sizes, int n_in,
                              void* d_out, int out_size, void* d_ws, size_t ws_size,
                              hipStream_t stream) {
  (void)in_sizes; (void)n_in; (void)out_size; (void)ws_size;
  const float* q    = (const float*)d_in[0];
  const float* rel  = (const float*)d_in[1];
  const float* Wq   = (const float*)d_in[2];
  const float* bq   = (const float*)d_in[3];
  const float* Wk   = (const float*)d_in[4];
  const float* bk   = (const float*)d_in[5];
  const float* Wv   = (const float*)d_in[6];
  const float* bv   = (const float*)d_in[7];
  const float* Wo   = (const float*)d_in[8];
  const float* bo   = (const float*)d_in[9];
  const float* ln_g = (const float*)d_in[10];
  const float* ln_b = (const float*)d_in[11];
  const int*   pidx = (const int*)d_in[13];   // position_indices (512x512 i32)
  float* out = (float*)d_out;

  char* ws = (char*)d_ws;
  bf16* xnb  = (bf16*)(ws + OFF_XNB);   // also reused as proj-out buffer
  bf16* Wqb  = (bf16*)(ws + OFF_WQB);
  bf16* Wkb  = (bf16*)(ws + OFF_WKB);
  bf16* Wvb  = (bf16*)(ws + OFF_WVB);
  bf16* Wob  = (bf16*)(ws + OFF_WOB);
  bf16* relb = (bf16*)(ws + OFF_RELB);
  bf16* Qs   = (bf16*)(ws + OFF_QS);
  bf16* Kb   = (bf16*)(ws + OFF_KB);
  bf16* Vb   = (bf16*)(ws + OFF_VB);
  bf16* qps  = (bf16*)(ws + OFF_QPS);
  bf16* kpb  = (bf16*)(ws + OFF_KPB);
  bf16* cpb  = (bf16*)(ws + OFF_CPB);
  bf16* pcb  = (bf16*)(ws + OFF_PCB);
  bf16* ctxb = (bf16*)(ws + OFF_CTX);

  // 0) precision staging
  ln_rows_kernel<<<SEQ * BB, 256, 0, stream>>>(q, xnb);
  cvt_bf16_kernel<<<(HH * HH) / 256, 256, 0, stream>>>(Wq, Wqb, HH * HH);
  cvt_bf16_kernel<<<(HH * HH) / 256, 256, 0, stream>>>(Wk, Wkb, HH * HH);
  cvt_bf16_kernel<<<(HH * HH) / 256, 256, 0, stream>>>(Wv, Wvb, HH * HH);
  cvt_bf16_kernel<<<(HH * HH) / 256, 256, 0, stream>>>(Wo, Wob, HH * HH);
  relpad_kernel<<<(64 * HH) / 256, 256, 0, stream>>>(rel, relb);

  // 1) projections via TDM-tiled GEMM (SCALE folded into Q and qp)
  dim3 ggrid(SEQ * BB / 128, HH / 128);
  gemm_tdm_kernel<<<ggrid, 256, 0, stream>>>(xnb, Wqb, bq, Qs, ATT_SCALE);
  gemm_tdm_kernel<<<ggrid, 256, 0, stream>>>(xnb, Wkb, bk, Kb, 1.0f);
  gemm_tdm_kernel<<<ggrid, 256, 0, stream>>>(xnb, Wvb, bv, Vb, 1.0f);
  gemm_bias_kernel<<<64 / 16, 256, 0, stream>>>(relb, Wqb, bq, qps, ATT_SCALE);
  gemm_bias_kernel<<<64 / 16, 256, 0, stream>>>(relb, Wkb, bk, kpb, 1.0f);

  // 2) relative-position projections per head
  cppc_gemm_kernel<<<dim3((SEQ * BB) / 16, 2), 256, 0, stream>>>(Qs, kpb, cpb);
  cppc_gemm_kernel<<<dim3((SEQ * BB) / 16, 2), 256, 0, stream>>>(Kb, qps, pcb);

  // 3) attention
  attn_kernel<<<dim3(SEQ / 64, NHH, BB), 128, 0, stream>>>(Qs, Kb, Vb, cpb, pcb, pidx, ctxb);

  // 4) output projection (TDM-tiled, into reused xnb buffer) + affine LN
  gemm_tdm_kernel<<<ggrid, 256, 0, stream>>>(ctxb, Wob, bo, xnb, 1.0f);
  ln_affine_kernel<<<SEQ * BB, 256, 0, stream>>>(xnb, ln_g, ln_b, out);
}
